// NodeDecoder_81690277970345
// MI455X (gfx1250) — compile-verified, matrix operand order checked by reading, not run on hardware
//
#include <hip/hip_runtime.h>
#include <hip/hip_bf16.h>
#include <math.h>

// ---- problem constants (match reference) ----
#define Hd     128
#define Vn     64
#define An     4096
#define NATTR  2048
#define Rn     (An - NATTR)          // 2048
#define PER_V  (NATTR + Rn * (Vn-1)) // 131072
#define TOTAL  (Vn * PER_V)          // 8388608

typedef __attribute__((ext_vector_type(16))) _Float16 v16h;
typedef __attribute__((ext_vector_type(8)))  _Float16 v8h;
typedef __attribute__((ext_vector_type(8)))  float    v8f;
typedef __attribute__((ext_vector_type(4)))  float    f4;

// ---------------------------------------------------------------------------
// gh = whh @ g + bhh   [384]
__global__ void k_gh(const float* __restrict__ whh, const float* __restrict__ bhh,
                     const float* __restrict__ g, float* __restrict__ gh) {
    int j = blockIdx.x * blockDim.x + threadIdx.x;
    if (j < 3 * Hd) {
        float s = bhh[j];
        const float* w = whh + j * Hd;
        #pragma unroll 8
        for (int k = 0; k < Hd; ++k) s += w[k] * g[k];
        gh[j] = s;
    }
}

// ---------------------------------------------------------------------------
// Per variable node: var1 MLP prob, GRU state, P_v = aor_W1b@state + aor_b1,
// Q_v = var2_W1a@local.  One block per v, 128 threads.
__global__ void k_var1(const float* __restrict__ emb, const float* __restrict__ g,
                       const int* __restrict__ v1locs,
                       const float* __restrict__ wih, const float* __restrict__ bih,
                       const float* __restrict__ gh,
                       const float* __restrict__ v1W1, const float* __restrict__ v1b1,
                       const float* __restrict__ v1W2, const float* __restrict__ v1b2,
                       const float* __restrict__ v1W3, const float* __restrict__ v1b3,
                       const float* __restrict__ aorW1, const float* __restrict__ aorb1,
                       const float* __restrict__ var2W1,
                       float* __restrict__ var1p, float* __restrict__ Pv, float* __restrict__ Qv) {
    __shared__ float sx[2 * Hd];
    __shared__ float st[Hd];
    __shared__ float sh1[Hd];
    __shared__ float sh2[Hd / 2];
    const int v = blockIdx.x, t = threadIdx.x;
    const int row = v1locs[v];
    sx[t]      = emb[(size_t)row * Hd + t];
    sx[Hd + t] = g[t];
    __syncthreads();

    // GRU gates (r,z,n), hidden = g
    float gi0 = bih[t], gi1 = bih[Hd + t], gi2 = bih[2 * Hd + t];
    #pragma unroll 4
    for (int k = 0; k < Hd; ++k) {
        float x = sx[k];
        gi0 += wih[(size_t)t * Hd + k] * x;
        gi1 += wih[(size_t)(Hd + t) * Hd + k] * x;
        gi2 += wih[(size_t)(2 * Hd + t) * Hd + k] * x;
    }
    float r = 1.f / (1.f + __expf(-(gi0 + gh[t])));
    float z = 1.f / (1.f + __expf(-(gi1 + gh[Hd + t])));
    float n = tanhf(gi2 + r * gh[2 * Hd + t]);
    st[t] = (1.f - z) * n + z * sx[Hd + t];

    // var1 MLP layer 1
    float h = v1b1[t];
    #pragma unroll 4
    for (int k = 0; k < 2 * Hd; ++k) h += v1W1[(size_t)t * 2 * Hd + k] * sx[k];
    sh1[t] = fmaxf(h, 0.f);
    __syncthreads();
    if (t < Hd / 2) {
        float h2 = v1b2[t];
        #pragma unroll 4
        for (int k = 0; k < Hd; ++k) h2 += v1W2[(size_t)t * Hd + k] * sh1[k];
        sh2[t] = fmaxf(h2, 0.f);
    }
    __syncthreads();
    if (t == 0) {
        float s = v1b3[0];
        for (int k = 0; k < Hd / 2; ++k) s += v1W3[k] * sh2[k];
        var1p[v] = s;
    }
    // P_v (carries aor_b1), Q_v (b folded into Q_r)
    float pv = aorb1[t], qv = 0.f;
    #pragma unroll 4
    for (int k = 0; k < Hd; ++k) {
        pv += aorW1[(size_t)t * 2 * Hd + Hd + k] * st[k];
        qv += var2W1[(size_t)t * 2 * Hd + k] * sx[k];
    }
    Pv[v * Hd + t] = pv;
    Qv[v * Hd + t] = qv;
}

// ---------------------------------------------------------------------------
// rela_states for aor rows [NATTR, An) then Q_r = var2_W1b@state + var2_b1.
__global__ void k_rela(const float* __restrict__ emb, const float* __restrict__ g,
                       const int* __restrict__ aorlocs,
                       const float* __restrict__ wih, const float* __restrict__ bih,
                       const float* __restrict__ gh,
                       const float* __restrict__ var2W1, const float* __restrict__ var2b1,
                       float* __restrict__ Qr) {
    __shared__ float sx[Hd];
    __shared__ float st[Hd];
    const int ri = blockIdx.x, t = threadIdx.x;
    const int row = aorlocs[NATTR + ri];
    sx[t] = emb[(size_t)row * Hd + t];
    __syncthreads();
    float gi0 = bih[t], gi1 = bih[Hd + t], gi2 = bih[2 * Hd + t];
    #pragma unroll 4
    for (int k = 0; k < Hd; ++k) {
        float x = sx[k];
        gi0 += wih[(size_t)t * Hd + k] * x;
        gi1 += wih[(size_t)(Hd + t) * Hd + k] * x;
        gi2 += wih[(size_t)(2 * Hd + t) * Hd + k] * x;
    }
    float r = 1.f / (1.f + __expf(-(gi0 + gh[t])));
    float z = 1.f / (1.f + __expf(-(gi1 + gh[Hd + t])));
    float n = tanhf(gi2 + r * gh[2 * Hd + t]);
    st[t] = (1.f - z) * n + z * g[t];
    __syncthreads();
    float q = var2b1[t];
    #pragma unroll 4
    for (int k = 0; k < Hd; ++k) q += var2W1[(size_t)t * 2 * Hd + Hd + k] * st[k];
    Qr[(size_t)ri * Hd + t] = q;
}

// ---------------------------------------------------------------------------
// P_a = aor_W1a @ local_aor   (bias lives in P_v)
__global__ void k_pa(const float* __restrict__ emb, const int* __restrict__ aorlocs,
                     const float* __restrict__ aorW1, float* __restrict__ Pa) {
    __shared__ float sx[Hd];
    const int a = blockIdx.x, t = threadIdx.x;
    sx[t] = emb[(size_t)aorlocs[a] * Hd + t];
    __syncthreads();
    float p = 0.f;
    #pragma unroll 4
    for (int k = 0; k < Hd; ++k) p += aorW1[(size_t)t * 2 * Hd + k] * sx[k];
    Pa[(size_t)a * Hd + t] = p;
}

// ---------------------------------------------------------------------------
// f32 -> f16 weight conversion for the WMMA layer-2 GEMMs (64x128 each)
__global__ void k_cvt(const float* __restrict__ aorW2, const float* __restrict__ var2W2,
                      _Float16* __restrict__ ha, _Float16* __restrict__ hv) {
    int i = blockIdx.x * 256 + threadIdx.x;
    if (i < 64 * Hd) {
        ha[i] = (_Float16)aorW2[i];
        hv[i] = (_Float16)var2W2[i];
    }
}

// ---------------------------------------------------------------------------
// Fused pairwise MLP tail:  scores[outer, inner] =
//   w3 . relu( W2 @ relu(Pout[outer] + Pin[inner]) + b2 ) + b3
// Layer-2 on v_wmma_f32_16x16x32_f16: tile = 16 inner rows, 4 K-steps, 4 N-tiles.
// C-fragment epilogue fuses relu+b2 and the layer-3 dot via shfl_xor over N lanes.
__global__ __launch_bounds__(256) void
k_pair(const float* __restrict__ Pout, const float* __restrict__ Pin,
       const _Float16* __restrict__ W2h, const float* __restrict__ b2,
       const float* __restrict__ w3, const float* __restrict__ b3,
       float* __restrict__ out, int nInner) {
    const int outer = blockIdx.x;
    const int lane  = threadIdx.x & 31;
    const int wave  = threadIdx.x >> 5;
    const int m     = lane & 15;   // row / col within 16
    const int o     = lane >> 4;   // half-wave group
    const int nTiles = nInner >> 4;
    const int stride = gridDim.y * 8;

    // B fragments: W2 as K x N, laid out per documented 16-bit B layout.
    v16h B[4][4];
    #pragma unroll
    for (int kt = 0; kt < 4; ++kt)
        #pragma unroll
        for (int nt = 0; nt < 4; ++nt) {
            const _Float16* p = W2h + (size_t)(nt * 16 + m) * Hd + kt * 32 + o * 8;
            v8h lo = *(const v8h*)p;
            v8h hi = *(const v8h*)(p + 16);
            v16h b;
            #pragma unroll
            for (int h = 0; h < 8; ++h) { b[h] = lo[h]; b[8 + h] = hi[h]; }
            B[kt][nt] = b;
        }
    float b2v[4], w3v[4];
    #pragma unroll
    for (int nt = 0; nt < 4; ++nt) { b2v[nt] = b2[nt * 16 + m]; w3v[nt] = w3[nt * 16 + m]; }
    const float bias3 = b3[0];
    const float* po = Pout + (size_t)outer * Hd;

    for (int tile = blockIdx.y * 8 + wave; tile < nTiles; tile += stride) {
        const float* pi = Pin + (size_t)(tile * 16 + m) * Hd;
        __builtin_prefetch(Pin + (size_t)((tile + stride) * 16 + m) * Hd, 0, 1);

        // A fragments: h1 = relu(Pin_row + Pout), packed per 16-bit A layout.
        v16h A[4];
        #pragma unroll
        for (int kt = 0; kt < 4; ++kt) {
            const int kb = kt * 32 + o * 8;
            f4 x0 = *(const f4*)(pi + kb);      f4 y0 = *(const f4*)(po + kb);
            f4 x1 = *(const f4*)(pi + kb + 4);  f4 y1 = *(const f4*)(po + kb + 4);
            f4 x2 = *(const f4*)(pi + kb + 16); f4 y2 = *(const f4*)(po + kb + 16);
            f4 x3 = *(const f4*)(pi + kb + 20); f4 y3 = *(const f4*)(po + kb + 20);
            v16h a;
            #pragma unroll
            for (int h = 0; h < 4; ++h) {
                a[h]      = (_Float16)fmaxf(x0[h] + y0[h], 0.f);
                a[4 + h]  = (_Float16)fmaxf(x1[h] + y1[h], 0.f);
                a[8 + h]  = (_Float16)fmaxf(x2[h] + y2[h], 0.f);
                a[12 + h] = (_Float16)fmaxf(x3[h] + y3[h], 0.f);
            }
            A[kt] = a;
        }

        v8f acc[4] = {};
        #pragma unroll
        for (int nt = 0; nt < 4; ++nt)
            #pragma unroll
            for (int kt = 0; kt < 4; ++kt)
                acc[nt] = __builtin_amdgcn_wmma_f32_16x16x32_f16(
                    false, A[kt], false, B[kt][nt], (short)0, acc[nt], false, false);

        // Layer 3: per row r (m = r + 8*o), s = sum_n relu(c+b2)*w3
        #pragma unroll
        for (int r = 0; r < 8; ++r) {
            float s = 0.f;
            #pragma unroll
            for (int nt = 0; nt < 4; ++nt)
                s += fmaxf(acc[nt][r] + b2v[nt], 0.f) * w3v[nt];
            s += __shfl_xor(s, 1);
            s += __shfl_xor(s, 2);
            s += __shfl_xor(s, 4);
            s += __shfl_xor(s, 8);
            if (m == 0)
                out[(size_t)outer * nInner + tile * 16 + r + 8 * o] = s + bias3;
        }
    }
}

// ---------------------------------------------------------------------------
// Streaming epilogue: 33.5 MB of coalesced writes, reads hit L2.
__global__ void k_out(const float* __restrict__ var1p, const float* __restrict__ aorP,
                      const float* __restrict__ v2s, float* __restrict__ out) {
    int e = blockIdx.x * 256 + threadIdx.x;
    if (e >= TOTAL) return;
    int v = e >> 17;               // PER_V = 131072
    int p = e & (PER_V - 1);
    float val = var1p[v];
    if (p < NATTR) {
        val += aorP[v * An + p];
    } else {
        int q = p - NATTR;
        int r = q / (Vn - 1);
        int j = q - r * (Vn - 1);
        int vj = j + (j >= v ? 1 : 0);
        val += aorP[v * An + NATTR + r] + v2s[r * Vn + vj];
    }
    out[e] = val;
}

// ---------------------------------------------------------------------------
extern "C" void kernel_launch(void* const* d_in, const int* in_sizes, int n_in,
                              void* d_out, int out_size, void* d_ws, size_t ws_size,
                              hipStream_t stream) {
    const float* emb   = (const float*)d_in[0];
    const float* g     = (const float*)d_in[1];
    const float* v1W1  = (const float*)d_in[2];
    const float* v1b1  = (const float*)d_in[3];
    const float* v1W2  = (const float*)d_in[4];
    const float* v1b2  = (const float*)d_in[5];
    const float* v1W3  = (const float*)d_in[6];
    const float* v1b3  = (const float*)d_in[7];
    const float* aorW1 = (const float*)d_in[8];
    const float* aorb1 = (const float*)d_in[9];
    const float* aorW2 = (const float*)d_in[10];
    const float* aorb2 = (const float*)d_in[11];
    const float* aorW3 = (const float*)d_in[12];
    const float* aorb3 = (const float*)d_in[13];
    const float* v2W1  = (const float*)d_in[14];
    const float* v2b1  = (const float*)d_in[15];
    const float* v2W2  = (const float*)d_in[16];
    const float* v2b2  = (const float*)d_in[17];
    const float* v2W3  = (const float*)d_in[18];
    const float* v2b3  = (const float*)d_in[19];
    const float* wih   = (const float*)d_in[20];
    const float* whh   = (const float*)d_in[21];
    const float* bih   = (const float*)d_in[22];
    const float* bhh   = (const float*)d_in[23];
    const int* v1locs  = (const int*)d_in[24];
    const int* aorlocs = (const int*)d_in[25];
    (void)in_sizes; (void)n_in; (void)out_size; (void)ws_size;

    float* ws = (float*)d_ws;
    float* gh    = ws;                    // 384
    float* var1p = gh + 384;              // 64
    float* Pv    = var1p + 64;            // 64*128
    float* Qv    = Pv + Vn * Hd;          // 64*128
    float* Qr    = Qv + Vn * Hd;          // 2048*128
    float* Pa    = Qr + Rn * Hd;          // 4096*128
    float* aorP  = Pa + An * Hd;          // 64*4096
    float* v2s   = aorP + Vn * An;        // 2048*64
    _Float16* W2ha = (_Float16*)(v2s + Rn * Vn);  // 64*128 halves
    _Float16* W2hv = W2ha + 64 * Hd;

    k_cvt <<<32, 256, 0, stream>>>(aorW2, v2W2, W2ha, W2hv);
    k_gh  <<<3, 128, 0, stream>>>(whh, bhh, g, gh);
    k_var1<<<Vn, Hd, 0, stream>>>(emb, g, v1locs, wih, bih, gh,
                                  v1W1, v1b1, v1W2, v1b2, v1W3, v1b3,
                                  aorW1, aorb1, v2W1, var1p, Pv, Qv);
    k_rela<<<Rn, Hd, 0, stream>>>(emb, g, aorlocs, wih, bih, gh, v2W1, v2b1, Qr);
    k_pa  <<<An, Hd, 0, stream>>>(emb, aorlocs, aorW1, Pa);

    // aor_probs[v,a]: outer=v (64), inner=a (4096): 256 tiles, 64 wave-slots x 4 tiles
    k_pair<<<dim3(Vn, 8), 256, 0, stream>>>(Pv, Pa, W2ha, aorb2, aorW3, aorb3, aorP, An);
    // var2_scores[r,v]: outer=r (2048), inner=v (64): 4 tiles
    k_pair<<<dim3(Rn, 1), 256, 0, stream>>>(Qr, Qv, W2hv, v2b2, v2W3, v2b3, v2s, Vn);

    k_out <<<(TOTAL + 255) / 256, 256, 0, stream>>>(var1p, aorP, v2s, (float*)d_out);
}
